// OptimizedAttention_39290360824086
// MI455X (gfx1250) — compile-verified, matrix operand order checked by reading, not run on hardware
//
#include <hip/hip_runtime.h>

#define B_DIM 2
#define S_DIM 2048
#define D_DIM 4096
#define NH 32
#define NKV 8
#define HD 128
#define QKV_OUT 6144

typedef __attribute__((ext_vector_type(16))) __bf16 v16bf;
typedef __attribute__((ext_vector_type(8)))  float  v8f;
typedef __attribute__((ext_vector_type(4)))  unsigned int u32x4;
typedef __attribute__((ext_vector_type(4)))  float  f32x4;

// gcc-style int4 vector matching the async-LDS builtin's parameter type
typedef int v4i __attribute__((vector_size(16)));
typedef __attribute__((address_space(1))) v4i* gptr_v4i;   // global (__device__)
typedef __attribute__((address_space(3))) v4i* lptr_v4i;   // LDS (__shared__)

union FragAB { v16bf v; u32x4 q[2]; };

#if defined(__HIP_DEVICE_COMPILE__) && defined(__gfx1250__)
#  if __has_builtin(__builtin_amdgcn_global_load_async_to_lds_b128)
#    define HAVE_ASYNC_LDS 1
#  endif
#  if __has_builtin(__builtin_amdgcn_s_wait_asynccnt)
#    define HAVE_WAIT_ASYNC 1
#  endif
#endif

#if defined(HAVE_WAIT_ASYNC)
#  define WAIT_ASYNC(n) __builtin_amdgcn_s_wait_asynccnt(n)
#elif defined(HAVE_ASYNC_LDS)
#  define WAIT_ASYNC(n) asm volatile("s_wait_asynccnt %0" :: "i"(n) : "memory")
#else
#  define WAIT_ASYNC(n) do {} while (0)
#endif

__device__ __forceinline__ unsigned short f32_to_bf16(float f) {
  union { float f; unsigned int u; } cv; cv.f = f;
  unsigned int u = cv.u;
  u += 0x7FFFu + ((u >> 16) & 1u);   // round-to-nearest-even
  return (unsigned short)(u >> 16);
}
__device__ __forceinline__ float bf16_to_f32(unsigned short h) {
  union { unsigned int u; float f; } cv; cv.u = ((unsigned int)h) << 16;
  return cv.f;
}

// ---------------------------------------------------------------- cast kernel
__global__ __launch_bounds__(256) void cast_f32_bf16_kernel(
    const float* __restrict__ src, unsigned short* __restrict__ dst, int n) {
  int i = (blockIdx.x * 256 + threadIdx.x) * 8;
  if (i + 7 < n) {
    f32x4 a = *(const f32x4*)(src + i);
    f32x4 b = *(const f32x4*)(src + i + 4);
    u32x4 o;
    o.x = (unsigned)f32_to_bf16(a.x) | ((unsigned)f32_to_bf16(a.y) << 16);
    o.y = (unsigned)f32_to_bf16(a.z) | ((unsigned)f32_to_bf16(a.w) << 16);
    o.z = (unsigned)f32_to_bf16(b.x) | ((unsigned)f32_to_bf16(b.y) << 16);
    o.w = (unsigned)f32_to_bf16(b.z) | ((unsigned)f32_to_bf16(b.w) << 16);
    *(u32x4*)(dst + i) = o;
  }
}

// ------------------------------------------------------- bf16 WMMA GEMM
// C[M,N] = A[M,K] * B[N,K]^T ; A,B bf16 row-major (K contiguous), C fp32 or bf16
// Workgroup tile 128x128, 8 waves, each wave 32x64 (8 WMMA accumulators).
// Double-buffered LDS: async copies for tile k+1 overlap WMMAs on tile k.
__global__ __launch_bounds__(256) void gemm_bf16_wmma_kernel(
    const unsigned short* __restrict__ A,
    const unsigned short* __restrict__ Bw,
    float* __restrict__ Cf, unsigned short* __restrict__ Cb,
    int M, int N, int K) {
  __shared__ unsigned short As[2][128][40];   // 128x32 tile, padded, x2 buffers
  __shared__ unsigned short Bs[2][128][40];

  const int tid  = threadIdx.x;
  const int lane = tid & 31;
  const int wv   = tid >> 5;
  const int wRow = (wv & 3) * 32;          // 4 strips of 32 rows
  const int wCol = (wv >> 2) * 64;         // 2 strips of 64 cols
  const int blockM = blockIdx.y * 128;
  const int blockN = blockIdx.x * 128;

  const int lrow    = lane & 15;
  const int halfSel = (lane < 16) ? 0 : 8; // K offset per 16-bit A/B layout

  v8f acc[2][4];
  for (int h = 0; h < 2; ++h)
    for (int s = 0; s < 4; ++s)
      for (int r = 0; r < 8; ++r) acc[h][s][r] = 0.0f;

  // staging: 128 rows x 32 cols each; 256 threads x 16 elems (2x b128) per matrix
  const int sRow = tid >> 1;               // 0..127
  const int sCol = (tid & 1) * 16;         // 0 or 16

  const unsigned short* aPtr = A  + (size_t)(blockM + sRow) * K + sCol;
  const unsigned short* bPtr = Bw + (size_t)(blockN + sRow) * K + sCol;

  const int ksteps = K >> 5;

  auto stage = [&](int kt, int buf) {
    const unsigned short* ap = aPtr + kt * 32;
    const unsigned short* bp = bPtr + kt * 32;
#if defined(HAVE_ASYNC_LDS)
    __builtin_amdgcn_global_load_async_to_lds_b128(
        (gptr_v4i)(void*)(ap),     (lptr_v4i)(void*)&As[buf][sRow][sCol], 0, 0);
    __builtin_amdgcn_global_load_async_to_lds_b128(
        (gptr_v4i)(void*)(ap + 8), (lptr_v4i)(void*)&As[buf][sRow][sCol + 8], 0, 0);
    __builtin_amdgcn_global_load_async_to_lds_b128(
        (gptr_v4i)(void*)(bp),     (lptr_v4i)(void*)&Bs[buf][sRow][sCol], 0, 0);
    __builtin_amdgcn_global_load_async_to_lds_b128(
        (gptr_v4i)(void*)(bp + 8), (lptr_v4i)(void*)&Bs[buf][sRow][sCol + 8], 0, 0);
#else
    u32x4 a0 = *(const u32x4*)(ap);
    u32x4 a1 = *(const u32x4*)(ap + 8);
    u32x4 b0 = *(const u32x4*)(bp);
    u32x4 b1 = *(const u32x4*)(bp + 8);
    *(u32x4*)&As[buf][sRow][sCol]     = a0;
    *(u32x4*)&As[buf][sRow][sCol + 8] = a1;
    *(u32x4*)&Bs[buf][sRow][sCol]     = b0;
    *(u32x4*)&Bs[buf][sRow][sCol + 8] = b1;
#endif
  };

  stage(0, 0);

  for (int kt = 0; kt < ksteps; ++kt) {
    const int cur = kt & 1;
    if (kt + 1 < ksteps) {
      stage(kt + 1, cur ^ 1);           // async copies for next tile in flight
      WAIT_ASYNC(4);                    // wait only for the CURRENT tile's copies
      if (kt + 2 < ksteps) {
        __builtin_prefetch(aPtr + (kt + 2) * 32, 0, 1);
        __builtin_prefetch(bPtr + (kt + 2) * 32, 0, 1);
      }
    } else {
      WAIT_ASYNC(0);
    }
    __syncthreads();

    FragAB a0, a1;
    a0.q[0] = *(const u32x4*)&As[cur][wRow + lrow][halfSel];
    a0.q[1] = *(const u32x4*)&As[cur][wRow + lrow][halfSel + 16];
    a1.q[0] = *(const u32x4*)&As[cur][wRow + 16 + lrow][halfSel];
    a1.q[1] = *(const u32x4*)&As[cur][wRow + 16 + lrow][halfSel + 16];
#pragma unroll
    for (int s = 0; s < 4; ++s) {
      FragAB bf;
      const int col = wCol + s * 16 + lrow;
      bf.q[0] = *(const u32x4*)&Bs[cur][col][halfSel];
      bf.q[1] = *(const u32x4*)&Bs[cur][col][halfSel + 16];
      acc[0][s] = __builtin_amdgcn_wmma_f32_16x16x32_bf16(
          false, a0.v, false, bf.v, (short)0, acc[0][s], false, false);
      acc[1][s] = __builtin_amdgcn_wmma_f32_16x16x32_bf16(
          false, a1.v, false, bf.v, (short)0, acc[1][s], false, false);
    }
    __syncthreads();   // all waves done reading buf[cur] before it is re-staged
  }

  for (int h = 0; h < 2; ++h) {
    for (int s = 0; s < 4; ++s) {
      const int col = blockN + wCol + s * 16 + lrow;
#pragma unroll
      for (int r = 0; r < 8; ++r) {
        const int row = blockM + wRow + h * 16 + r + ((lane < 16) ? 0 : 8);
        const float v = acc[h][s][r];
        if (Cf) Cf[(size_t)row * N + col] = v;
        else    Cb[(size_t)row * N + col] = f32_to_bf16(v);
      }
    }
  }
}

// ---------------------------------------------------------------- RoPE/split
__global__ __launch_bounds__(256) void rope_split_kernel(
    const unsigned short* __restrict__ qkv,
    unsigned short* __restrict__ Qb,
    unsigned short* __restrict__ Kb,
    unsigned short* __restrict__ Vb) {
  int idx = blockIdx.x * 256 + threadIdx.x;
  const int i = idx & 63;        // pair index 0..63
  int t = idx >> 6;
  const int slot = t % 48;       // 32 q heads + 8 k heads + 8 v heads
  t /= 48;
  const int s = t % S_DIM;
  const int b = t / S_DIM;
  if (b >= B_DIM) return;

  const unsigned short* row = qkv + ((size_t)(b * S_DIM + s)) * QKV_OUT + slot * HD;
  const float x1 = bf16_to_f32(row[i]);
  const float x2 = bf16_to_f32(row[i + 64]);

  if (slot < NH + NKV) {  // rotary on Q and K
    const float ang = (float)s * __powf(500000.0f, -(float)(2 * i) * (1.0f / 128.0f));
    const float c = __cosf(ang), sn = __sinf(ang);
    unsigned short* o;
    if (slot < NH)
      o = Qb + (((size_t)(b * NH + slot)) * S_DIM + s) * HD;
    else
      o = Kb + (((size_t)(b * NKV + (slot - NH))) * S_DIM + s) * HD;
    o[i]      = f32_to_bf16(x1 * c - x2 * sn);
    o[i + 64] = f32_to_bf16(x2 * c + x1 * sn);
  } else {
    unsigned short* o = Vb + (((size_t)(b * NKV + (slot - NH - NKV))) * S_DIM + s) * HD;
    o[i]      = row[i];
    o[i + 64] = row[i + 64];
  }
}

// ---------------------------------------------------- flash attention (WMMA)
__global__ __launch_bounds__(64) void flash_attn_wmma_kernel(
    const unsigned short* __restrict__ Qb,
    const unsigned short* __restrict__ Kb,
    const unsigned short* __restrict__ Vb,
    unsigned short* __restrict__ Ao) {
  __shared__ unsigned short Ks[2][32][HD];   // K block, row-major (async-filled)
  __shared__ unsigned short Vt[2][HD][40];   // V block, transposed + padded
  __shared__ unsigned short Ps[2][16][40];   // P tile (C-layout -> A-layout hop)

  const int lane = threadIdx.x & 31;
  const int wv   = threadIdx.x >> 5;
  const int tile = blockIdx.x * 2 + wv;
  const int qt  = tile & (S_DIM / 16 - 1);
  const int h   = (tile >> 7) & (NH - 1);
  const int b   = tile >> 12;
  const int hkv = h >> 2;

  const int lrow    = lane & 15;
  const int halfSel = (lane < 16) ? 0 : 8;
  const int rowOff  = (lane < 16) ? 0 : 8;

  const unsigned short* qbase =
      Qb + (((size_t)(b * NH + h)) * S_DIM + qt * 16 + lrow) * HD;
  FragAB qa[4];
#pragma unroll
  for (int s = 0; s < 4; ++s) {
    qa[s].q[0] = *(const u32x4*)(qbase + s * 32 + halfSel);
    qa[s].q[1] = *(const u32x4*)(qbase + s * 32 + halfSel + 16);
  }

  float m[8], l[8];
  v8f O[8];
  for (int r = 0; r < 8; ++r) { m[r] = -1e30f; l[r] = 0.0f; }
  for (int c = 0; c < 8; ++c)
    for (int r = 0; r < 8; ++r) O[c][r] = 0.0f;

  const int qlo = qt * 16;
  const int nkb = ((qlo + 15) >> 5) + 1;
  const unsigned short* kbase = Kb + ((size_t)(b * NKV + hkv)) * S_DIM * HD;
  const unsigned short* vbase = Vb + ((size_t)(b * NKV + hkv)) * S_DIM * HD;
  const float scale = 0.088388347648318447f;  // 1/sqrt(128)

  for (int kb2 = 0; kb2 < nkb; ++kb2) {
    const int kpos0 = kb2 * 32;
    {   // stage K via async copies, V via register transpose (overlapped)
      const unsigned short* kr = kbase + (size_t)(kpos0 + lane) * HD;
      const unsigned short* vr = vbase + (size_t)(kpos0 + lane) * HD;
#if defined(HAVE_ASYNC_LDS)
#pragma unroll
      for (int c = 0; c < 16; ++c)
        __builtin_amdgcn_global_load_async_to_lds_b128(
            (gptr_v4i)(void*)(kr + c * 8), (lptr_v4i)(void*)&Ks[wv][lane][c * 8], 0, 0);
#else
#pragma unroll
      for (int c = 0; c < 16; ++c)
        *(u32x4*)&Ks[wv][lane][c * 8] = *(const u32x4*)(kr + c * 8);
#endif
#pragma unroll
      for (int c = 0; c < 16; ++c) {
        union { u32x4 q; unsigned short u[8]; } vvu;
        vvu.q = *(const u32x4*)(vr + c * 8);
#pragma unroll
        for (int j = 0; j < 8; ++j) Vt[wv][c * 8 + j][lane] = vvu.u[j];
      }
      WAIT_ASYNC(0);                    // per-wave counter: K tile resident
    }
    asm volatile("s_wait_dscnt 0x0" ::: "memory");

    v8f s0, s1;
    for (int r = 0; r < 8; ++r) { s0[r] = 0.0f; s1[r] = 0.0f; }
#pragma unroll
    for (int s = 0; s < 4; ++s) {
      FragAB k0f, k1f;
      k0f.q[0] = *(const u32x4*)&Ks[wv][lrow][s * 32 + halfSel];
      k0f.q[1] = *(const u32x4*)&Ks[wv][lrow][s * 32 + halfSel + 16];
      k1f.q[0] = *(const u32x4*)&Ks[wv][16 + lrow][s * 32 + halfSel];
      k1f.q[1] = *(const u32x4*)&Ks[wv][16 + lrow][s * 32 + halfSel + 16];
      s0 = __builtin_amdgcn_wmma_f32_16x16x32_bf16(false, qa[s].v, false, k0f.v, (short)0, s0, false, false);
      s1 = __builtin_amdgcn_wmma_f32_16x16x32_bf16(false, qa[s].v, false, k1f.v, (short)0, s1, false, false);
    }

#pragma unroll
    for (int r = 0; r < 8; ++r) {
      const int qrow = qlo + r + rowOff;
      float a0 = s0[r] * scale;
      float a1 = s1[r] * scale;
      if (kpos0 + lrow > qrow)      a0 = -3.0e38f;
      if (kpos0 + 16 + lrow > qrow) a1 = -3.0e38f;
      float t = fmaxf(a0, a1);                       // row-max across 16 lanes
      t = fmaxf(t, __shfl_xor(t, 1, 32));
      t = fmaxf(t, __shfl_xor(t, 2, 32));
      t = fmaxf(t, __shfl_xor(t, 4, 32));
      t = fmaxf(t, __shfl_xor(t, 8, 32));
      const float mn   = fmaxf(m[r], t);
      const float corr = __expf(m[r] - mn);
      m[r] = mn;
      const float p0 = __expf(a0 - mn);
      const float p1 = __expf(a1 - mn);
      float rs = p0 + p1;
      rs += __shfl_xor(rs, 1, 32);
      rs += __shfl_xor(rs, 2, 32);
      rs += __shfl_xor(rs, 4, 32);
      rs += __shfl_xor(rs, 8, 32);
      l[r] = l[r] * corr + rs;
#pragma unroll
      for (int c = 0; c < 8; ++c) O[c][r] *= corr;
      Ps[wv][r + rowOff][lrow]      = f32_to_bf16(p0);  // C-layout -> LDS
      Ps[wv][r + rowOff][16 + lrow] = f32_to_bf16(p1);
    }
    asm volatile("s_wait_dscnt 0x0" ::: "memory");

    FragAB pa;                                           // LDS -> A-layout
    pa.q[0] = *(const u32x4*)&Ps[wv][lrow][halfSel];
    pa.q[1] = *(const u32x4*)&Ps[wv][lrow][halfSel + 16];
#pragma unroll
    for (int c = 0; c < 8; ++c) {
      FragAB vbf;
      vbf.q[0] = *(const u32x4*)&Vt[wv][c * 16 + lrow][halfSel];
      vbf.q[1] = *(const u32x4*)&Vt[wv][c * 16 + lrow][halfSel + 16];
      O[c] = __builtin_amdgcn_wmma_f32_16x16x32_bf16(false, pa.v, false, vbf.v, (short)0, O[c], false, false);
    }
  }

#pragma unroll
  for (int r = 0; r < 8; ++r) {
    const float inv = 1.0f / l[r];
    const int qrow  = qlo + r + rowOff;
    unsigned short* dst = Ao + ((size_t)(b * S_DIM + qrow)) * D_DIM + h * HD;
#pragma unroll
    for (int c = 0; c < 8; ++c)
      dst[c * 16 + lrow] = f32_to_bf16(O[c][r] * inv);
  }
}

// ------------------------------------------------------------------- launch
extern "C" void kernel_launch(void* const* d_in, const int* in_sizes, int n_in,
                              void* d_out, int out_size, void* d_ws, size_t ws_size,
                              hipStream_t stream) {
  (void)in_sizes; (void)n_in; (void)out_size; (void)ws_size;
  const float* x    = (const float*)d_in[0];
  const float* Wqkv = (const float*)d_in[1];
  const float* Wo   = (const float*)d_in[2];
  float* out = (float*)d_out;

  const size_t n_x    = (size_t)B_DIM * S_DIM * D_DIM;
  const size_t n_wqkv = (size_t)QKV_OUT * D_DIM;
  const size_t n_wo   = (size_t)D_DIM * D_DIM;
  const size_t n_qkv  = (size_t)B_DIM * S_DIM * QKV_OUT;
  const size_t n_q    = (size_t)B_DIM * NH * S_DIM * HD;
  const size_t n_kv   = (size_t)B_DIM * NKV * S_DIM * HD;

  char* ws = (char*)d_ws;
  size_t off = 0;
  auto ALLOC = [&](size_t elems) {
    unsigned short* p = (unsigned short*)(ws + off);
    off += ((elems * 2 + 255) / 256) * 256;
    return p;
  };
  unsigned short* xb    = ALLOC(n_x);
  unsigned short* wqkvb = ALLOC(n_wqkv);
  unsigned short* wob   = ALLOC(n_wo);
  unsigned short* qkvb  = ALLOC(n_qkv);
  unsigned short* Qbuf  = ALLOC(n_q);
  unsigned short* Kbuf  = ALLOC(n_kv);
  unsigned short* Vbuf  = ALLOC(n_kv);
  unsigned short* Abuf  = ALLOC(n_x);

  cast_f32_bf16_kernel<<<dim3((unsigned)(n_x / 2048)),    256, 0, stream>>>(x,    xb,    (int)n_x);
  cast_f32_bf16_kernel<<<dim3((unsigned)(n_wqkv / 2048)), 256, 0, stream>>>(Wqkv, wqkvb, (int)n_wqkv);
  cast_f32_bf16_kernel<<<dim3((unsigned)(n_wo / 2048)),   256, 0, stream>>>(Wo,   wob,   (int)n_wo);

  gemm_bf16_wmma_kernel<<<dim3(QKV_OUT / 128, (B_DIM * S_DIM) / 128), 256, 0, stream>>>(
      xb, wqkvb, nullptr, qkvb, B_DIM * S_DIM, QKV_OUT, D_DIM);

  rope_split_kernel<<<dim3((B_DIM * S_DIM * 48 * 64) / 256), 256, 0, stream>>>(
      qkvb, Qbuf, Kbuf, Vbuf);

  flash_attn_wmma_kernel<<<dim3((B_DIM * NH * (S_DIM / 16)) / 2), 64, 0, stream>>>(
      Qbuf, Kbuf, Vbuf, Abuf);

  gemm_bf16_wmma_kernel<<<dim3(D_DIM / 128, (B_DIM * S_DIM) / 128), 256, 0, stream>>>(
      Abuf, wob, out, nullptr, B_DIM * S_DIM, D_DIM, D_DIM);
}
